// Codebook_5488968204908
// MI455X (gfx1250) — matrix operand order repaired and checked by reading, not compiled, not run on hardware
//
#include <hip/hip_runtime.h>

typedef __bf16 bf16_t;
typedef bf16_t v16bf __attribute__((ext_vector_type(16)));
typedef float  v8f   __attribute__((ext_vector_type(8)));

#define NUM_VECTORS 8192
#define LATENT      256
#define M_TOTAL     16384      // B*H*W = 16*32*32
#define OUT_IMG     4194304    // 16*256*32*32
#define A_STRIDE    264        // 256 + 8 pad (bf16 elems) -> 528B row stride, conflict-free b128 reads
#define MT          128        // rows per block in GEMM kernels
#define NT          32         // codes per B tile in argmin kernel (2 N-subtiles per wave)

// ---------------- helpers ----------------

__device__ __forceinline__ void bf16_split(float f, unsigned short& hi, unsigned short& lo) {
    unsigned u = __float_as_uint(f);
    unsigned r = (u + 0x7FFFu + ((u >> 16) & 1u)) >> 16;   // RNE truncate to bf16
    hi = (unsigned short)r;
    float fh = __uint_as_float(r << 16);
    float fl = f - fh;
    unsigned u2 = __float_as_uint(fl);
    lo = (unsigned short)((u2 + 0x7FFFu + ((u2 >> 16) & 1u)) >> 16);
}

__device__ __forceinline__ unsigned f2ord(float f) {
    unsigned u = __float_as_uint(f);
    return (u & 0x80000000u) ? ~u : (u | 0x80000000u);     // monotone float->uint
}

// load a 16-elem bf16 fragment as two 16B chunks (ds_load_b128 when p is LDS)
struct alignas(16) FragBits { int4 a, b; };
__device__ __forceinline__ v16bf ld_frag(const unsigned short* p0, const unsigned short* p1) {
    FragBits u;
    u.a = *(const int4*)p0;
    u.b = *(const int4*)p1;
    return __builtin_bit_cast(v16bf, u);
}

#define WMMA_BF16(A, B, C) \
    __builtin_amdgcn_wmma_f32_16x16x32_bf16(false, (A), false, (B), (short)0, (C), false, false)

// 16B async copy global -> LDS (gfx1250 GLOBAL_LOAD_ASYNC_TO_LDS_B128, tracked by ASYNCcnt)
typedef int nat_int4 __attribute__((vector_size(16)));                      // clang native v4i32
typedef __attribute__((address_space(3))) nat_int4 lds_nat_int4;
__device__ __forceinline__ void async_cp16(unsigned short* lds_dst, const unsigned short* gsrc) {
    __builtin_amdgcn_global_load_async_to_lds_b128(
        (nat_int4*)gsrc,
        (lds_nat_int4*)lds_dst,
        0, 0);
}

// ---------------- prep kernels ----------------

// x: (16,256,32,32) -> row-major (bhw, c) split into bf16 hi/lo, via 32x32 LDS transpose tiles
__global__ __launch_bounds__(256) void k_split_x(const float* __restrict__ x,
                                                 unsigned short* __restrict__ xh,
                                                 unsigned short* __restrict__ xl,
                                                 float* __restrict__ loss_slot) {
    __shared__ unsigned short th[32][33];
    __shared__ unsigned short tl[32][33];
    int rt = blockIdx.x;          // 0..511 : bhw tile
    int ct = blockIdx.y;          // 0..7   : channel tile
    int tx = threadIdx.x;         // 0..31
    int ty = threadIdx.y;         // 0..7
    if (rt == 0 && ct == 0 && tx == 0 && ty == 0) *loss_slot = 0.0f;   // zero loss accumulator
    int row0 = rt * 32;
    int b    = row0 >> 10;
    int hw0  = row0 & 1023;
    #pragma unroll
    for (int j = 0; j < 4; ++j) {
        int cl = ty + 8 * j;
        int c  = ct * 32 + cl;
        float v = x[(size_t)b * 262144 + (size_t)c * 1024 + (hw0 + tx)];  // coalesced over hw
        unsigned short h, l;
        bf16_split(v, h, l);
        th[tx][cl] = h;
        tl[tx][cl] = l;
    }
    __syncthreads();
    #pragma unroll
    for (int j = 0; j < 4; ++j) {
        int rl  = ty + 8 * j;
        int row = row0 + rl;
        int c   = ct * 32 + tx;                                            // coalesced over c
        xh[(size_t)row * 256 + c] = th[rl][tx];
        xl[(size_t)row * 256 + c] = tl[rl][tx];
    }
}

__global__ __launch_bounds__(256) void k_split_e(const float* __restrict__ e,
                                                 unsigned short* __restrict__ eh,
                                                 unsigned short* __restrict__ el, int n) {
    int t = blockIdx.x * 256 + threadIdx.x;
    if (t < n) {
        unsigned short h, l;
        bf16_split(e[t], h, l);
        eh[t] = h; el[t] = l;
    }
}

__global__ __launch_bounds__(256) void k_enorm(const float* __restrict__ e,
                                               float* __restrict__ en2) {
    int wid = threadIdx.x >> 5, lane = threadIdx.x & 31;
    int row = blockIdx.x * 8 + wid;
    const float* p = e + (size_t)row * 256;
    float s = 0.0f;
    #pragma unroll
    for (int j = 0; j < 8; ++j) { float v = p[lane + 32 * j]; s += v * v; }
    #pragma unroll
    for (int off = 16; off > 0; off >>= 1) s += __shfl_down(s, off, 32);
    if (lane == 0) en2[row] = s;
}

// emb_value (k,n) -> transposed (n,k) bf16 hi/lo so B-fragments are contiguous in K
__global__ __launch_bounds__(256) void k_split_vT(const float* __restrict__ v,
                                                  unsigned short* __restrict__ vh,
                                                  unsigned short* __restrict__ vl) {
    __shared__ unsigned short th[32][33];
    __shared__ unsigned short tl[32][33];
    int kt = blockIdx.x, nt = blockIdx.y;      // 8 x 8 tiles
    int tx = threadIdx.x, ty = threadIdx.y;
    #pragma unroll
    for (int j = 0; j < 4; ++j) {
        int kl = ty + 8 * j;
        int k  = kt * 32 + kl;
        int n  = nt * 32 + tx;                 // coalesced over n
        unsigned short h, l;
        bf16_split(v[(size_t)k * 256 + n], h, l);
        th[kl][tx] = h;
        tl[kl][tx] = l;
    }
    __syncthreads();
    #pragma unroll
    for (int j = 0; j < 4; ++j) {
        int nl = ty + 8 * j;
        int n  = nt * 32 + nl;
        int k  = kt * 32 + tx;                 // coalesced over k
        vh[(size_t)n * 256 + k] = th[tx][nl];
        vl[(size_t)n * 256 + k] = tl[tx][nl];
    }
}

// ---------------- main kernel: distances + argmin ----------------
// block = 256 threads (8 waves); each wave owns 16 rows and two 16-col subtiles per B tile.
// B tiles (32 codes, hi+lo) are double-buffered in LDS, filled by async global->LDS copies.
__global__ __launch_bounds__(256) void k_argmin(const unsigned short* __restrict__ xh,
                                                const unsigned short* __restrict__ xl,
                                                const unsigned short* __restrict__ eh,
                                                const unsigned short* __restrict__ el,
                                                const float* __restrict__ en2g,
                                                int* __restrict__ idxi,
                                                float* __restrict__ idxf) {
    __shared__ unsigned short sAh[MT * A_STRIDE];
    __shared__ unsigned short sAl[MT * A_STRIDE];
    __shared__ unsigned short sBh[2][NT * A_STRIDE];
    __shared__ unsigned short sBl[2][NT * A_STRIDE];
    __shared__ unsigned long long sBest[MT];

    int tid   = threadIdx.x;
    int rbase = blockIdx.x * MT;

    // kick off async fill of B tile 0 (8 x 16B per thread: 4 hi + 4 lo chunks)
    #pragma unroll
    for (int j = 0; j < 4; ++j) {
        int u = tid + 256 * j;                 // 0..1023 : 32 rows x 32 k-chunks
        int r = u >> 5, kc = u & 31;
        async_cp16(&sBh[0][r * A_STRIDE + kc * 8], &eh[(size_t)r * 256 + kc * 8]);
        async_cp16(&sBl[0][r * A_STRIDE + kc * 8], &el[(size_t)r * 256 + kc * 8]);
    }

    // stage 128 x-rows (hi+lo) into LDS, 16B chunks, coalesced
    for (int u = tid; u < MT * 32; u += 256) {
        int r = u >> 5, kc = u & 31;
        *(int4*)&sAh[r * A_STRIDE + kc * 8] = *(const int4*)&xh[(size_t)(rbase + r) * 256 + kc * 8];
        *(int4*)&sAl[r * A_STRIDE + kc * 8] = *(const int4*)&xl[(size_t)(rbase + r) * 256 + kc * 8];
    }
    if (tid < MT) sBest[tid] = ~0ull;

    int wid = tid >> 5, lane = tid & 31;
    int lane15 = lane & 15, hi = lane >> 4;
    // A fragment base: lane = row M, one contiguous 16-elem run per K32 chunk (lanes>=16 shifted by 16)
    const unsigned short* pAh = &sAh[(wid * 16 + lane15) * A_STRIDE + hi * 16];
    const unsigned short* pAl = &sAl[(wid * 16 + lane15) * A_STRIDE + hi * 16];
    // B fragment offsets: lane = col N, one contiguous 16-elem run per K32 chunk
    int bofs0 = lane15 * A_STRIDE + hi * 16;          // subtile 0: codes nb + 0..15
    int bofs1 = (16 + lane15) * A_STRIDE + hi * 16;   // subtile 1: codes nb + 16..31

    float best[8];
    int   bidx[8];
    #pragma unroll
    for (int i = 0; i < 8; ++i) { best[i] = 3.4e38f; bidx[i] = 0; }

    int cur = 0;
    for (int nb = 0; nb < NUM_VECTORS; nb += NT) {
        if (nb + NT < NUM_VECTORS) {
            // issue async fill of the next tile into the other buffer
            int nxt = cur ^ 1;
            #pragma unroll
            for (int j = 0; j < 4; ++j) {
                int u = tid + 256 * j;
                int r = u >> 5, kc = u & 31;
                async_cp16(&sBh[nxt][r * A_STRIDE + kc * 8],
                           &eh[(size_t)(nb + NT + r) * 256 + kc * 8]);
                async_cp16(&sBl[nxt][r * A_STRIDE + kc * 8],
                           &el[(size_t)(nb + NT + r) * 256 + kc * 8]);
            }
            __builtin_amdgcn_s_wait_asynccnt(8);   // the 8 just-issued may remain in flight
        } else {
            __builtin_amdgcn_s_wait_asynccnt(0);   // drain for the final tile
        }
        __syncthreads();                           // publish tile `cur` to all waves

        const unsigned short* pBh0 = &sBh[cur][bofs0];
        const unsigned short* pBl0 = &sBl[cur][bofs0];
        const unsigned short* pBh1 = &sBh[cur][bofs1];
        const unsigned short* pBl1 = &sBl[cur][bofs1];

        v8f acc0 = {}, acc1 = {};
        #pragma unroll
        for (int kc = 0; kc < 8; ++kc) {
            int kb = kc * 32;
            v16bf ah  = ld_frag(pAh + kb, pAh + kb + 8);
            v16bf al  = ld_frag(pAl + kb, pAl + kb + 8);
            v16bf bh0 = ld_frag(pBh0 + kb, pBh0 + kb + 8);
            v16bf bl0 = ld_frag(pBl0 + kb, pBl0 + kb + 8);
            v16bf bh1 = ld_frag(pBh1 + kb, pBh1 + kb + 8);
            v16bf bl1 = ld_frag(pBl1 + kb, pBl1 + kb + 8);
            acc0 = WMMA_BF16(ah, bh0, acc0);   // bf16x3: hi*hi + hi*lo + lo*hi, fp32 accumulate
            acc0 = WMMA_BF16(ah, bl0, acc0);
            acc0 = WMMA_BF16(al, bh0, acc0);
            acc1 = WMMA_BF16(ah, bh1, acc1);
            acc1 = WMMA_BF16(ah, bl1, acc1);
            acc1 = WMMA_BF16(al, bh1, acc1);
        }

        float en2a = en2g[nb + lane15];
        float en2b = en2g[nb + 16 + lane15];
        #pragma unroll
        for (int i = 0; i < 8; ++i) {
            float s0 = en2a - 2.0f * acc0[i];       // ||e||^2 - 2 x.e  (||x||^2 const per row)
            if (s0 < best[i]) { best[i] = s0; bidx[i] = nb + lane15; }
            float s1 = en2b - 2.0f * acc1[i];
            if (s1 < best[i]) { best[i] = s1; bidx[i] = nb + 16 + lane15; }
        }
        __syncthreads();   // all reads of tile `cur` done before it is refilled next iter
        cur ^= 1;
    }

    // merge 16 column-slots per row via ds_min_u64 on packed (dist, idx): ties -> smallest idx
    #pragma unroll
    for (int i = 0; i < 8; ++i) {
        unsigned long long key =
            ((unsigned long long)f2ord(best[i]) << 32) | (unsigned)bidx[i];
        atomicMin(&sBest[wid * 16 + i + 8 * hi], key);
    }
    __syncthreads();
    if (tid < MT) {
        int n = (int)(unsigned)(sBest[tid] & 0xFFFFFFFFull);
        idxi[rbase + tid] = n;
        idxf[rbase + tid] = (float)n;
    }
}

// ---------------- value kernel: xq = emb_key[idx] @ emb_value, outputs + loss ----------------
__global__ __launch_bounds__(256) void k_value(const unsigned short* __restrict__ eh,
                                               const unsigned short* __restrict__ el,
                                               const unsigned short* __restrict__ vh,
                                               const unsigned short* __restrict__ vl,
                                               const int* __restrict__ idxi,
                                               const float* __restrict__ x,
                                               float* __restrict__ out_st,
                                               float* __restrict__ out_grad,
                                               float* __restrict__ loss) {
    __shared__ unsigned short sAh[MT * A_STRIDE];
    __shared__ unsigned short sAl[MT * A_STRIDE];
    __shared__ int   sIdx[MT];
    __shared__ float sT[8][16][17];

    int tid   = threadIdx.x;
    int rbase = blockIdx.x * MT;
    if (tid < MT) sIdx[tid] = idxi[rbase + tid];
    __syncthreads();

    // gather selected code rows into LDS
    for (int u = tid; u < MT * 32; u += 256) {
        int r = u >> 5, kc = u & 31;
        int code = sIdx[r];
        *(int4*)&sAh[r * A_STRIDE + kc * 8] = *(const int4*)&eh[(size_t)code * 256 + kc * 8];
        *(int4*)&sAl[r * A_STRIDE + kc * 8] = *(const int4*)&el[(size_t)code * 256 + kc * 8];
    }
    __syncthreads();

    int wid = tid >> 5, lane = tid & 31;
    int lane15 = lane & 15, hi = lane >> 4;
    const unsigned short* pAh = &sAh[(wid * 16 + lane15) * A_STRIDE + hi * 16];
    const unsigned short* pAl = &sAl[(wid * 16 + lane15) * A_STRIDE + hi * 16];

    int b      = rbase >> 10;                  // block never crosses a batch boundary (128 | 1024)
    int hwbase = (rbase & 1023) + wid * 16;
    float lsum = 0.0f;

    for (int nc = 0; nc < 16; ++nc) {
        int nb = nc * 16;
        const unsigned short* pBh = &vh[(size_t)(nb + lane15) * 256 + hi * 16];  // vT: L2/WGP$ resident
        const unsigned short* pBl = &vl[(size_t)(nb + lane15) * 256 + hi * 16];
        v8f acc = {};
        #pragma unroll
        for (int kc = 0; kc < 8; ++kc) {
            int kb = kc * 32;
            v16bf ah = ld_frag(pAh + kb, pAh + kb + 8);
            v16bf al = ld_frag(pAl + kb, pAl + kb + 8);
            v16bf bh = ld_frag(pBh + kb, pBh + kb + 8);
            v16bf bl = ld_frag(pBl + kb, pBl + kb + 8);
            acc = WMMA_BF16(ah, bh, acc);
            acc = WMMA_BF16(ah, bl, acc);
            acc = WMMA_BF16(al, bh, acc);
        }
        // stage tile to LDS so global writes are coalesced along hw
        #pragma unroll
        for (int i = 0; i < 8; ++i) sT[wid][i + 8 * hi][lane15] = acc[i];
        __syncthreads();
        #pragma unroll
        for (int it = 0; it < 8; ++it) {
            int cl = hi + it * 2;
            int m  = lane15;
            float val = sT[wid][m][cl];
            size_t addr = ((size_t)b << 18) + ((size_t)(nb + cl) << 10) + (hwbase + m);
            out_st[addr]   = val;              // straight-through value == xq
            out_grad[addr] = val;
            float d = val - x[addr];
            lsum += d * d;
        }
        __syncthreads();
    }

    #pragma unroll
    for (int off = 16; off > 0; off >>= 1) lsum += __shfl_down(lsum, off, 32);
    if (lane == 0) atomicAdd(loss, lsum * (1.25f / (float)OUT_IMG));  // (1+beta)*mean
}

// ---------------- launcher ----------------

extern "C" void kernel_launch(void* const* d_in, const int* in_sizes, int n_in,
                              void* d_out, int out_size, void* d_ws, size_t ws_size,
                              hipStream_t stream) {
    const float* x  = (const float*)d_in[0];   // (16,256,32,32)
    const float* ek = (const float*)d_in[1];   // (8192,256)
    const float* ev = (const float*)d_in[2];   // (256,256)

    float* out      = (float*)d_out;
    float* out_st   = out;
    float* out_grad = out + OUT_IMG;
    float* out_idx  = out + 2 * (size_t)OUT_IMG;
    float* out_loss = out + 2 * (size_t)OUT_IMG + M_TOTAL;

    char* ws = (char*)d_ws;
    unsigned short* xh  = (unsigned short*)ws;  ws += (size_t)M_TOTAL * 256 * 2;
    unsigned short* xl  = (unsigned short*)ws;  ws += (size_t)M_TOTAL * 256 * 2;
    unsigned short* eh  = (unsigned short*)ws;  ws += (size_t)NUM_VECTORS * 256 * 2;
    unsigned short* el  = (unsigned short*)ws;  ws += (size_t)NUM_VECTORS * 256 * 2;
    unsigned short* vTh = (unsigned short*)ws;  ws += (size_t)256 * 256 * 2;
    unsigned short* vTl = (unsigned short*)ws;  ws += (size_t)256 * 256 * 2;
    float*          en2 = (float*)ws;           ws += (size_t)NUM_VECTORS * 4;
    int*            idxi = (int*)ws;            ws += (size_t)M_TOTAL * 4;

    k_split_x <<<dim3(512, 8), dim3(32, 8), 0, stream>>>(x, xh, xl, out_loss);
    k_split_e <<<(NUM_VECTORS * LATENT + 255) / 256, 256, 0, stream>>>(ek, eh, el, NUM_VECTORS * LATENT);
    k_enorm   <<<NUM_VECTORS / 8, 256, 0, stream>>>(ek, en2);
    k_split_vT<<<dim3(8, 8), dim3(32, 8), 0, stream>>>(ev, vTh, vTl);
    k_argmin  <<<M_TOTAL / MT, 256, 0, stream>>>(xh, xl, eh, el, en2, idxi, out_idx);
    k_value   <<<M_TOTAL / MT, 256, 0, stream>>>(eh, el, vTh, vTl, idxi, x, out_st, out_grad, out_loss);
}
